// EdgeBlock_36069135352227
// MI455X (gfx1250) — compile-verified, matrix operand order checked by reading, not run on hardware
//
#include <hip/hip_runtime.h>
#include <hip/hip_bf16.h>
#include <math.h>

typedef __attribute__((ext_vector_type(16))) __bf16 v16bf;
typedef __attribute__((ext_vector_type(8)))  __bf16 v8bf;
typedef __attribute__((ext_vector_type(8)))  float  v8f;

#define DN 128
#define DE 128
#define K3 640    // 3*DN + 2*DE
#define NOUT 256  // 2*DE
#define YSTRIDE 260   // f32 LDS row stride for [32 x 256] GEMM result (bank-skewed)
#define ESTRIDE 132   // f32 LDS row stride for [32 x 128] GLU result
#define X2STRIDE 136  // bf16 LDS row stride for c2 activation tile (16B-aligned, skewed)
#define X3STRIDE 648  // bf16 LDS row stride for c3 activation tile (1296B = 81*16)

// Build a 16-element bf16 WMMA fragment from two contiguous 8-element (16B) chunks.
// Per ISA A/B 16-bit layout: lane elements 0..7 -> k = kb + half*8 + 0..7,
// elements 8..15 -> k = kb + 16 + half*8 + 0..7  (both contiguous runs).
__device__ __forceinline__ v16bf frag16(const __bf16* p) {
  union { v16bf v; v8bf h[2]; } u;
  u.h[0] = *(const v8bf*)(p);
  u.h[1] = *(const v8bf*)(p + 16);
  return u.v;
}

__device__ __forceinline__ float sigm(float x) { return 1.0f / (1.0f + __expf(-x)); }

// Async global->LDS copy of 16 bytes per lane (CDNA5, tracked by ASYNCcnt).
__device__ __forceinline__ void async_b128(unsigned lds_off, const void* gaddr) {
  asm volatile("global_load_async_to_lds_b128 %0, %1, off"
               :: "v"(lds_off), "v"(gaddr) : "memory");
}
__device__ __forceinline__ void wait_async0() {
  asm volatile("s_wait_asynccnt 0" ::: "memory");
}

// ---------------------------------------------------------------------------
// f32 -> bf16 embedding tables (node/edge), float4 reads.
// ---------------------------------------------------------------------------
__global__ void k_conv_emb(const float* __restrict__ src, __bf16* __restrict__ dst,
                           int n4) {
  int o = blockIdx.x * blockDim.x + threadIdx.x;
  if (o >= n4) return;
  float4 v = *(const float4*)(src + (size_t)o * 4);
  __bf16* d = dst + (size_t)o * 4;
  d[0] = (__bf16)v.x; d[1] = (__bf16)v.y; d[2] = (__bf16)v.z; d[3] = (__bf16)v.w;
}

// Weight convert: w[k][n] (f32, row-major [K,N]) -> wT[n][k] (bf16) so a
// B-fragment column (fixed n, contiguous k) is two 16-byte global loads.
__global__ void k_conv_w(const float* __restrict__ w, __bf16* __restrict__ wT,
                         int K, int N) {
  int o = blockIdx.x * blockDim.x + threadIdx.x;
  if (o >= K * N) return;
  int n = o / K, k = o % K;
  wT[o] = (__bf16)w[k * N + n];
}

// ---------------------------------------------------------------------------
// c2: per 32-edge tile: x = node[i]*node[j] (bf16 in, f32 product, bf16 tile);
// y = x @ w_c2 + b; LN(256); GLU; LN(128); write pre = edge + c2_emb into out.
// ---------------------------------------------------------------------------
__launch_bounds__(128)
__global__ void k_c2(const __bf16* __restrict__ node_bf,
                     const int* __restrict__ ii, const int* __restrict__ jj,
                     const __bf16* __restrict__ w2T,
                     const float* __restrict__ bias,
                     const float* __restrict__ g1, const float* __restrict__ b1,
                     const float* __restrict__ g2, const float* __restrict__ b2,
                     const float* __restrict__ edge,
                     float* __restrict__ out, int E) {
  __shared__ __align__(16) __bf16 xs[32 * X2STRIDE];
  __shared__ __align__(16) float  ys[32 * YSTRIDE];
  __shared__ __align__(16) float  es[32 * ESTRIDE];
  __shared__ float mean1[32], rstd1[32], mean2[32], rstd2[32];

  const int tid  = threadIdx.x;
  const int base = blockIdx.x * 32;

  // Gather + elementwise product -> bf16 tile [32 x 128]; 4 threads per row.
  {
    int r = tid >> 2, q = tid & 3;
    int e0 = base + r;
    if (e0 >= E) e0 = E - 1;
    const __bf16* pa = node_bf + (size_t)ii[e0] * DN;
    const __bf16* pb = node_bf + (size_t)jj[e0] * DN;
    for (int off = q * 8; off < DN; off += 32) {
      v8bf a8 = *(const v8bf*)(pa + off);
      v8bf b8 = *(const v8bf*)(pb + off);
      __bf16* d = xs + r * X2STRIDE + off;
#pragma unroll
      for (int u = 0; u < 8; ++u)
        d[u] = (__bf16)((float)a8[u] * (float)b8[u]);
    }
  }
  __syncthreads();

  // GEMM [32x128] @ [128x256] with v_wmma_f32_16x16x32_bf16.
  const int lane = tid & 31, wid = tid >> 5;
  const int mt = wid & 1, nh = wid >> 1;
  const int l16 = lane & 15, half = lane >> 4;
  v8f acc[8] = {};
  for (int kb = 0; kb < DN / 32; ++kb) {
    const __bf16* ap = xs + (mt * 16 + l16) * X2STRIDE + kb * 32 + half * 8;
    v16bf a = frag16(ap);
#pragma unroll
    for (int t = 0; t < 8; ++t) {
      const __bf16* bp = w2T + (size_t)(nh * 128 + t * 16 + l16) * DN + kb * 32 + half * 8;
      v16bf b = frag16(bp);
      acc[t] = __builtin_amdgcn_wmma_f32_16x16x32_bf16(
          false, a, false, b, (short)0, acc[t], false, false);
    }
  }
#pragma unroll
  for (int t = 0; t < 8; ++t) {
    int col = nh * 128 + t * 16 + l16;
    float bv = bias[col];
#pragma unroll
    for (int r = 0; r < 8; ++r) {
      int row = mt * 16 + half * 8 + r;   // C/D layout: VGPR r -> M = r (+8 hi lanes)
      ys[row * YSTRIDE + col] = acc[t][r] + bv;
    }
  }
  __syncthreads();

  // LN over 256 per row
  if (tid < 32) {
    float s = 0.f, ss = 0.f;
    for (int c = 0; c < NOUT; ++c) { float v = ys[tid * YSTRIDE + c]; s += v; ss += v * v; }
    float m = s * (1.0f / NOUT);
    mean1[tid] = m;
    rstd1[tid] = rsqrtf(ss * (1.0f / NOUT) - m * m + 1e-5f);
  }
  __syncthreads();

  // GLU: sigmoid(filter) * tanh(core)
  {
    int c = tid;
    float gF = g1[c], bF = b1[c], gC = g1[c + 128], bC = b1[c + 128];
    for (int r = 0; r < 32; ++r) {
      float m = mean1[r], rs = rstd1[r];
      float f  = (ys[r * YSTRIDE + c] - m) * rs * gF + bF;
      float cc = (ys[r * YSTRIDE + c + 128] - m) * rs * gC + bC;
      es[r * ESTRIDE + c] = sigm(f) * tanhf(cc);
    }
  }
  __syncthreads();

  // LN over 128 per row
  if (tid < 32) {
    float s = 0.f, ss = 0.f;
    for (int c = 0; c < DE; ++c) { float v = es[tid * ESTRIDE + c]; s += v; ss += v * v; }
    float m = s * (1.0f / DE);
    mean2[tid] = m;
    rstd2[tid] = rsqrtf(ss * (1.0f / DE) - m * m + 1e-5f);
  }
  __syncthreads();

  // pre-output: edge + c2_emb  (k_final adds LN(c3_sum) and applies tanh)
  {
    int c = tid;
    float g = g2[c], b = b2[c];
    for (int r = 0; r < 32; ++r) {
      int e0 = base + r;
      if (e0 < E) {
        size_t o = (size_t)e0 * DE + c;
        out[o] = edge[o] + (es[r * ESTRIDE + c] - mean2[r]) * rstd2[r] * g + b;
      }
    }
  }
}

// ---------------------------------------------------------------------------
// c3: per 32-triplet tile: x = concat(5 bf16 gathers) [32x640] copied into LDS
// with GLOBAL_LOAD_ASYNC_TO_LDS_B128; y = x @ w_c3 + b; LN(256); GLU; atomic
// scatter-add into c3_sum[index_ji].  f32 y-tile aliases the bf16 x-tile.
// ---------------------------------------------------------------------------
__launch_bounds__(128)
__global__ void k_c3(const __bf16* __restrict__ node_bf, const __bf16* __restrict__ edge_bf,
                     const int* __restrict__ idx_i, const int* __restrict__ idx_j,
                     const int* __restrict__ idx_k, const int* __restrict__ idx_ji,
                     const int* __restrict__ idx_kj,
                     const __bf16* __restrict__ w3T, const float* __restrict__ bias,
                     const float* __restrict__ g1, const float* __restrict__ b1,
                     float* __restrict__ c3_sum, int T) {
  __shared__ __align__(16) unsigned char smem[32 * X3STRIDE * 2]; // 41472 B
  __shared__ float mean1[32], rstd1[32];
  __bf16* xs = (__bf16*)smem;
  float*  ys = (float*)smem;   // 32*260*4 = 33280 B <= 41472 B (aliased)

  const int tid  = threadIdx.x;
  const int base = blockIdx.x * 32;

  // Async gather: each thread owns (row r, quarter q); 5 segments x 4 chunks of
  // 16B per thread, written straight to LDS (no VGPR round trip).
  {
    int r = tid >> 2, q = tid & 3;
    int t0 = base + r;
    if (t0 >= T) t0 = T - 1;
    const __bf16* p0 = node_bf + (size_t)idx_i[t0] * DN;
    const __bf16* p1 = node_bf + (size_t)idx_j[t0] * DN;
    const __bf16* p2 = node_bf + (size_t)idx_k[t0] * DN;
    const __bf16* p3 = edge_bf + (size_t)idx_ji[t0] * DE;
    const __bf16* p4 = edge_bf + (size_t)idx_kj[t0] * DE;
    // low 32 bits of a generic LDS pointer == DS byte offset (flat->LDS trunc)
    unsigned ldsrow = (unsigned)(unsigned long long)(const void*)xs
                      + (unsigned)(r * (X3STRIDE * 2) + q * 16);
#pragma unroll
    for (int seg = 0; seg < 5; ++seg) {
      const __bf16* bp = seg == 0 ? p0 : seg == 1 ? p1 : seg == 2 ? p2 : seg == 3 ? p3 : p4;
      const char* src = (const char*)bp + q * 16;
      unsigned dst = ldsrow + seg * 256;
#pragma unroll
      for (int ch = 0; ch < 4; ++ch)
        async_b128(dst + ch * 64, src + ch * 64);
    }
    wait_async0();
  }
  __syncthreads();

  const int lane = tid & 31, wid = tid >> 5;
  const int mt = wid & 1, nh = wid >> 1;
  const int l16 = lane & 15, half = lane >> 4;
  v8f acc[8] = {};
  for (int kb = 0; kb < K3 / 32; ++kb) {
    const __bf16* ap = xs + (mt * 16 + l16) * X3STRIDE + kb * 32 + half * 8;
    v16bf a = frag16(ap);
#pragma unroll
    for (int t = 0; t < 8; ++t) {
      const __bf16* bp = w3T + (size_t)(nh * 128 + t * 16 + l16) * K3 + kb * 32 + half * 8;
      v16bf b = frag16(bp);
      acc[t] = __builtin_amdgcn_wmma_f32_16x16x32_bf16(
          false, a, false, b, (short)0, acc[t], false, false);
    }
  }
  __syncthreads();   // all xs reads complete before aliased ys writes

#pragma unroll
  for (int t = 0; t < 8; ++t) {
    int col = nh * 128 + t * 16 + l16;
    float bv = bias[col];
#pragma unroll
    for (int r = 0; r < 8; ++r) {
      int row = mt * 16 + half * 8 + r;
      ys[row * YSTRIDE + col] = acc[t][r] + bv;
    }
  }
  __syncthreads();

  if (tid < 32) {
    float s = 0.f, ss = 0.f;
    for (int c = 0; c < NOUT; ++c) { float v = ys[tid * YSTRIDE + c]; s += v; ss += v * v; }
    float m = s * (1.0f / NOUT);
    mean1[tid] = m;
    rstd1[tid] = rsqrtf(ss * (1.0f / NOUT) - m * m + 1e-5f);
  }
  __syncthreads();

  {
    int c = tid;
    float gF = g1[c], bF = b1[c], gC = g1[c + 128], bC = b1[c + 128];
    for (int r = 0; r < 32; ++r) {
      int t0 = base + r;
      if (t0 >= T) break;
      float m = mean1[r], rs = rstd1[r];
      float f  = (ys[r * YSTRIDE + c] - m) * rs * gF + bF;
      float cc = (ys[r * YSTRIDE + c + 128] - m) * rs * gC + bC;
      float msg = sigm(f) * tanhf(cc);
      unsafeAtomicAdd(&c3_sum[(size_t)idx_ji[t0] * DE + c], msg);  // global_atomic_add_f32
    }
  }
}

// ---------------------------------------------------------------------------
// Final: out = tanh(pre + LN(c3_sum)) in place.  One wave32 per edge row,
// shuffle-reduced layer norm, float4 I/O.
// ---------------------------------------------------------------------------
__launch_bounds__(256)
__global__ void k_final(const float* __restrict__ c3s, const float* __restrict__ g,
                        const float* __restrict__ b, float* out, int E) {
  int wid = threadIdx.x >> 5, lane = threadIdx.x & 31;
  int row = blockIdx.x * 8 + wid;
  if (row >= E) return;
  size_t off = (size_t)row * DE + lane * 4;
  float4 v = *(const float4*)(c3s + off);
  float s = v.x + v.y + v.z + v.w;
#pragma unroll
  for (int d = 16; d > 0; d >>= 1) s += __shfl_xor(s, d, 32);
  float m = s * (1.0f / DE);
  float4 dv = make_float4(v.x - m, v.y - m, v.z - m, v.w - m);
  float q = dv.x * dv.x + dv.y * dv.y + dv.z * dv.z + dv.w * dv.w;
#pragma unroll
  for (int d = 16; d > 0; d >>= 1) q += __shfl_xor(q, d, 32);
  float rstd = rsqrtf(q * (1.0f / DE) + 1e-5f);
  int c = lane * 4;
  float4 pv = *(const float4*)(out + off);
  float4 gv = *(const float4*)(g + c);
  float4 bv = *(const float4*)(b + c);
  float4 o;
  o.x = tanhf(pv.x + dv.x * rstd * gv.x + bv.x);
  o.y = tanhf(pv.y + dv.y * rstd * gv.y + bv.y);
  o.z = tanhf(pv.z + dv.z * rstd * gv.z + bv.z);
  o.w = tanhf(pv.w + dv.w * rstd * gv.w + bv.w);
  *(float4*)(out + off) = o;
}

extern "C" void kernel_launch(void* const* d_in, const int* in_sizes, int n_in,
                              void* d_out, int out_size, void* d_ws, size_t ws_size,
                              hipStream_t stream) {
  const float* node   = (const float*)d_in[0];
  const float* edge   = (const float*)d_in[1];
  const int*   ii     = (const int*)d_in[2];
  const int*   jj     = (const int*)d_in[3];
  const int*   idx_i  = (const int*)d_in[4];
  const int*   idx_j  = (const int*)d_in[5];
  const int*   idx_k  = (const int*)d_in[6];
  const int*   idx_ji = (const int*)d_in[7];
  const int*   idx_kj = (const int*)d_in[8];
  const float* w_c2   = (const float*)d_in[9];
  const float* b_c2   = (const float*)d_in[10];
  const float* w_c3   = (const float*)d_in[11];
  const float* b_c3   = (const float*)d_in[12];
  const float* g_bn_c2  = (const float*)d_in[13];
  const float* be_bn_c2 = (const float*)d_in[14];
  const float* g_bn_c3  = (const float*)d_in[15];
  const float* be_bn_c3 = (const float*)d_in[16];
  const float* g_c2_2   = (const float*)d_in[17];
  const float* be_c2_2  = (const float*)d_in[18];
  const float* g_c3_2   = (const float*)d_in[19];
  const float* be_c3_2  = (const float*)d_in[20];

  const int N = in_sizes[0] / DN;  // 20000 nodes
  const int E = in_sizes[1] / DE;  // 400000 edges
  const int T = in_sizes[4];       // 500000 triplets

  // Workspace carve-up (all regions 16B aligned).
  float*  c3_sum  = (float*)d_ws;
  __bf16* node_bf = (__bf16*)(c3_sum + (size_t)E * DE);
  __bf16* edge_bf = node_bf + (size_t)N * DN;
  __bf16* w2T     = edge_bf + (size_t)E * DE;
  __bf16* w3T     = w2T + (size_t)DN * NOUT;

  hipMemsetAsync(c3_sum, 0, (size_t)E * DE * sizeof(float), stream);

  k_conv_emb<<<(N * DN / 4 + 255) / 256, 256, 0, stream>>>(node, node_bf, N * DN / 4);
  k_conv_emb<<<(E * DE / 4 + 255) / 256, 256, 0, stream>>>(edge, edge_bf, E * DE / 4);
  k_conv_w<<<(DN * NOUT + 255) / 256, 256, 0, stream>>>(w_c2, w2T, DN, NOUT);
  k_conv_w<<<(K3 * NOUT + 255) / 256, 256, 0, stream>>>(w_c3, w3T, K3, NOUT);

  k_c2<<<(E + 31) / 32, 128, 0, stream>>>(node_bf, ii, jj, w2T, b_c2,
                                          g_bn_c2, be_bn_c2, g_c2_2, be_c2_2,
                                          edge, (float*)d_out, E);
  k_c3<<<(T + 31) / 32, 128, 0, stream>>>(node_bf, edge_bf, idx_i, idx_j, idx_k,
                                          idx_ji, idx_kj, w3T, b_c3,
                                          g_bn_c3, be_bn_c3, c3_sum, T);
  k_final<<<(E + 7) / 8, 256, 0, stream>>>(c3_sum, g_c3_2, be_c3_2,
                                           (float*)d_out, E);
}
